// GCNModel_49933289783662
// MI455X (gfx1250) — matrix operand order repaired and checked by reading, not compiled, not run on hardware
//
#include <hip/hip_runtime.h>

// ---------------------------------------------------------------------------
// CDNA5 (gfx1250) GCN pipeline: bf16 WMMA GEMMs + coalesced edge scatter.
// ---------------------------------------------------------------------------

typedef __attribute__((ext_vector_type(16))) __bf16 v16bf;
typedef __attribute__((ext_vector_type(8)))  float  v8f;

__device__ __forceinline__ unsigned short f2bf(float f) {
  unsigned int u = __float_as_uint(f);
  if ((u & 0x7fffffffu) > 0x7f800000u)            // NaN -> quiet bf16 NaN
    return (unsigned short)((u >> 16) | 0x40u);
  return (unsigned short)((u + 0x7fffu + ((u >> 16) & 1u)) >> 16); // RNE
}

// ---- degree / norm -------------------------------------------------------
__global__ void k_deg(const int* __restrict__ dst, float* __restrict__ deg, int E) {
  int i = blockIdx.x * blockDim.x + threadIdx.x;
  int stride = gridDim.x * blockDim.x;
  for (; i < E; i += stride) atomicAdd(&deg[dst[i]], 1.0f);
}

__global__ void k_dinv(float* __restrict__ deg, int N) {
  int i = blockIdx.x * blockDim.x + threadIdx.x;
  if (i < N) deg[i] = rsqrtf(deg[i] + 1.0f);      // +1 = self loop; deg >= 1
}

// ---- f32 -> bf16 converts ------------------------------------------------
__global__ void k_f2bf(const float* __restrict__ in, unsigned short* __restrict__ out, int n) {
  int i = blockIdx.x * blockDim.x + threadIdx.x;
  if (i < n) out[i] = f2bf(in[i]);
}

// W [K,N] f32 row-major -> Wt [N,K] bf16 (so B-operand K values are contiguous)
__global__ void k_wt_bf(const float* __restrict__ W, unsigned short* __restrict__ Wt,
                        int K, int N) {
  int i = blockIdx.x * blockDim.x + threadIdx.x;
  if (i < K * N) {
    int k = i / N, n = i - k * N;
    Wt[(size_t)n * K + k] = f2bf(W[i]);
  }
}

// ---- bf16 WMMA GEMM: C[M,N] = A[M,K] * B[K,N], B given transposed --------
// Per-wave tile: 16 (M) x 64 (N); K stepped by 32.
__global__ void __launch_bounds__(256) k_gemm_wmma(
    const unsigned short* __restrict__ A,   // [M,K] bf16 row-major
    const unsigned short* __restrict__ Bt,  // [N,K] bf16 row-major (B^T)
    float* __restrict__ C, int M, int K, int N) {
  const int lane   = threadIdx.x & 31;
  const int wave   = (blockIdx.x * blockDim.x + threadIdx.x) >> 5;
  const int nChunk64 = N >> 6;
  const int mTile  = wave / nChunk64;
  const int nChunk = wave - mTile * nChunk64;
  if (mTile * 16 >= M) return;

  const int l15 = lane & 15;
  const int hi  = lane >> 4;                 // lane half selects K sub-block
  const unsigned short* aRow = A + (size_t)(mTile * 16 + l15) * K + hi * 8;

  v8f acc[4] = {};
  for (int k = 0; k < K; k += 32) {
    // A 16x32 bf16 layout: lanes 0-15 row m hold K {0..7,16..23}; lanes 16-31 {8..15,24..31}
    union { v16bf v; uint4 q[2]; } a;
    a.q[0] = *(const uint4*)(aRow + k);
    a.q[1] = *(const uint4*)(aRow + k + 16);
#pragma unroll
    for (int t = 0; t < 4; ++t) {
      // B 32x16 bf16 layout: lanes 0-15 col n hold K {0..7,16..23}; lanes 16-31 {8..15,24..31}
      const unsigned short* bRow =
          Bt + (size_t)(nChunk * 64 + t * 16 + l15) * K + k + hi * 8;
      union { v16bf v; uint4 q[2]; } b;
      b.q[0] = *(const uint4*)(bRow);
      b.q[1] = *(const uint4*)(bRow + 16);
      acc[t] = __builtin_amdgcn_wmma_f32_16x16x32_bf16(
          false, a.v, false, b.v, (short)0, acc[t], false, false);
    }
  }
  // C/D f32 layout: VGPR r -> (m = 8*hi + r, n = l15)
  const int mBase = mTile * 16 + hi * 8;
#pragma unroll
  for (int t = 0; t < 4; ++t) {
    float* cp = C + (size_t)mBase * N + nChunk * 64 + t * 16 + l15;
#pragma unroll
    for (int r = 0; r < 8; ++r) cp[(size_t)r * N] = acc[t][r];
  }
}

// ---- edge scatter: agg[dst] += h[src] * dinv[src]*dinv[dst] --------------
// One block per edge, thread = channel (fully coalesced gather + atomic add).
__global__ void k_scatter(const float* __restrict__ h, const int* __restrict__ src,
                          const int* __restrict__ dst, const float* __restrict__ dinv,
                          float* __restrict__ agg, int C) {
  int e = blockIdx.x;
  int s = src[e], d = dst[e];
  float nrm = dinv[s] * dinv[d];
  int c = threadIdx.x;
  atomicAdd(&agg[(size_t)d * C + c], h[(size_t)s * C + c] * nrm);
}

// ---- epilogue: self-loop + bias + relu (+ bf16 out and/or pool atomics) --
__global__ void k_post(const float* __restrict__ agg, const float* __restrict__ hlin,
                       const float* __restrict__ dinv, const float* __restrict__ bias,
                       unsigned short* __restrict__ obf,  // next-layer bf16 input (or null)
                       float* __restrict__ pool,          // graph pool sums (or null)
                       const int* __restrict__ batch, int C) {
  int i = blockIdx.x;
  int c = threadIdx.x;
  float dv = dinv[i];
  float v = agg[(size_t)i * C + c] + hlin[(size_t)i * C + c] * (dv * dv) + bias[c];
  v = fmaxf(v, 0.0f);
  if (obf)  obf[(size_t)i * C + c] = f2bf(v);
  if (pool) atomicAdd(&pool[(size_t)batch[i] * C + c], v);
}

__global__ void k_cnt(const int* __restrict__ batch, float* __restrict__ cnt, int N) {
  int i = blockIdx.x * blockDim.x + threadIdx.x;
  if (i < N) atomicAdd(&cnt[batch[i]], 1.0f);
}

// ---- head: out[g,r] = (pool[g,:]/max(cnt,1)) . Wr[:,r] + br[r] -----------
__global__ void k_head(const float* __restrict__ pool, const float* __restrict__ cnt,
                       const float* __restrict__ Wr, const float* __restrict__ br,
                       float* __restrict__ out, int G, int R, int C) {
  int idx = blockIdx.x * blockDim.x + threadIdx.x;
  if (idx >= G * R) return;
  int g = idx / R, r = idx - g * R;
  float inv = 1.0f / fmaxf(cnt[g], 1.0f);
  float acc = 0.0f;
  for (int k = 0; k < C; ++k)
    acc = fmaf(pool[(size_t)g * C + k] * inv, Wr[(size_t)k * R + r], acc);
  out[idx] = acc + br[r];
}

// ---------------------------------------------------------------------------
extern "C" void kernel_launch(void* const* d_in, const int* in_sizes, int n_in,
                              void* d_out, int out_size, void* d_ws, size_t ws_size,
                              hipStream_t stream) {
  const float* x     = (const float*)d_in[0];
  const int*   ei    = (const int*)  d_in[1];
  const int*   batch = (const int*)  d_in[2];
  const float* W1 = (const float*)d_in[3]; const float* b1 = (const float*)d_in[4];
  const float* W2 = (const float*)d_in[5]; const float* b2 = (const float*)d_in[6];
  const float* W3 = (const float*)d_in[7]; const float* b3 = (const float*)d_in[8];
  const float* Wr = (const float*)d_in[9]; const float* br = (const float*)d_in[10];
  float* out = (float*)d_out;

  const int Nn = in_sizes[2];            // 50000 nodes
  const int E  = in_sizes[1] / 2;        // 800000 edges
  const int CIN = 128, CH1 = 256, CH2 = 256, COUT = 128, R = 2001;
  const int G = out_size / R;            // 128 graphs
  const int* src = ei;
  const int* dst = ei + E;

  // workspace carve-out (256B aligned)
  char* ws = (char*)d_ws;
  size_t cur = 0;
  auto carve = [&](size_t bytes) -> void* {
    void* p = ws + cur;
    cur = (cur + bytes + 255) & ~(size_t)255;
    return p;
  };
  float*          dinv  = (float*)carve((size_t)Nn * 4);
  unsigned short* xb    = (unsigned short*)carve((size_t)Nn * 256 * 2); // bf16 activations
  float*          hlin  = (float*)carve((size_t)Nn * 256 * 4);
  float*          agg   = (float*)carve((size_t)Nn * 256 * 4);
  unsigned short* w1t   = (unsigned short*)carve((size_t)CIN * CH1 * 2);
  unsigned short* w2t   = (unsigned short*)carve((size_t)CH1 * CH2 * 2);
  unsigned short* w3t   = (unsigned short*)carve((size_t)CH2 * COUT * 2);
  float*          psum  = (float*)carve((size_t)G * COUT * 4);
  float*          pcnt  = (float*)carve((size_t)G * 4);

  // --- degrees & symmetric norm ---
  hipMemsetAsync(dinv, 0, (size_t)Nn * 4, stream);
  k_deg <<<1024, 256, 0, stream>>>(dst, dinv, E);
  k_dinv<<<(Nn + 255) / 256, 256, 0, stream>>>(dinv, Nn);

  // --- convert inputs/weights to bf16 ---
  k_f2bf <<<((size_t)Nn * CIN + 255) / 256, 256, 0, stream>>>(x, xb, Nn * CIN);
  k_wt_bf<<<(CIN * CH1 + 255) / 256, 256, 0, stream>>>(W1, w1t, CIN, CH1);
  k_wt_bf<<<(CH1 * CH2 + 255) / 256, 256, 0, stream>>>(W2, w2t, CH1, CH2);
  k_wt_bf<<<(CH2 * COUT + 255) / 256, 256, 0, stream>>>(W3, w3t, CH2, COUT);

  auto gemm = [&](const unsigned short* A, const unsigned short* Bt, float* C,
                  int M, int K, int N) {
    int waves  = (M / 16) * (N / 64);
    int blocks = (waves + 7) / 8;        // 8 waves (256 threads) per block
    k_gemm_wmma<<<blocks, 256, 0, stream>>>(A, Bt, C, M, K, N);
  };

  // --- layer 1: 128 -> 256 ---
  gemm(xb, w1t, hlin, Nn, CIN, CH1);
  hipMemsetAsync(agg, 0, (size_t)Nn * CH1 * 4, stream);
  k_scatter<<<E, CH1, 0, stream>>>(hlin, src, dst, dinv, agg, CH1);
  k_post<<<Nn, CH1, 0, stream>>>(agg, hlin, dinv, b1, xb, nullptr, batch, CH1);

  // --- layer 2: 256 -> 256 ---
  gemm(xb, w2t, hlin, Nn, CH1, CH2);
  hipMemsetAsync(agg, 0, (size_t)Nn * CH2 * 4, stream);
  k_scatter<<<E, CH2, 0, stream>>>(hlin, src, dst, dinv, agg, CH2);
  k_post<<<Nn, CH2, 0, stream>>>(agg, hlin, dinv, b2, xb, nullptr, batch, CH2);

  // --- layer 3: 256 -> 128, epilogue fuses the graph mean-pool sums ---
  gemm(xb, w3t, hlin, Nn, CH2, COUT);
  hipMemsetAsync(agg, 0, (size_t)Nn * COUT * 4, stream);
  hipMemsetAsync(psum, 0, (size_t)G * COUT * 4, stream);
  hipMemsetAsync(pcnt, 0, (size_t)G * 4, stream);
  k_scatter<<<E, COUT, 0, stream>>>(hlin, src, dst, dinv, agg, COUT);
  k_cnt<<<(Nn + 255) / 256, 256, 0, stream>>>(batch, pcnt, Nn);
  k_post<<<Nn, COUT, 0, stream>>>(agg, hlin, dinv, b3, nullptr, psum, batch, COUT);

  // --- readout head: [G,128] @ [128,2001] + br ---
  k_head<<<((size_t)G * R + 255) / 256, 256, 0, stream>>>(psum, pcnt, Wr, br, out, G, R, COUT);
}
